// QuantizeWrapper_90692529423065
// MI455X (gfx1250) — 8-bit matrix operand order repaired and checked by reading, compiled, not run on hardware
//
#include <hip/hip_runtime.h>

typedef __attribute__((ext_vector_type(8))) int v8i;
typedef __attribute__((ext_vector_type(4))) int v4i;
typedef __attribute__((address_space(1))) void as1_void;
typedef __attribute__((address_space(3))) void as3_void;
typedef __attribute__((address_space(1))) v4i as1_v4i;
typedef __attribute__((address_space(3))) v4i as3_v4i;

#define M_TOT 8192
#define N_TOT 8192
#define K_TOT 2048

#if defined(__has_builtin)
#  if __has_builtin(__builtin_amdgcn_global_load_async_to_lds_b128)
#    define USE_ASYNC_LDS 1
#  endif
#endif
#ifndef USE_ASYNC_LDS
#  define USE_ASYNC_LDS 0
#endif

__device__ __forceinline__ void wait_async_done() {
#if USE_ASYNC_LDS
#  if __has_builtin(__builtin_amdgcn_s_wait_asynccnt)
    __builtin_amdgcn_s_wait_asynccnt(0);
#  else
    asm volatile("s_wait_asynccnt 0x0" ::: "memory");
#  endif
#endif
}

__device__ __forceinline__ int quant1(float v, float inv_scale) {
    float q = rintf(v * inv_scale);
    q = fminf(fmaxf(q, -128.0f), 127.0f);
    return (int)q;
}

__global__ __launch_bounds__(256)
void quant_x_kernel(const float* __restrict__ x,
                    const float* __restrict__ in_scale,
                    signed char* __restrict__ qx, int n4) {
    int i = blockIdx.x * blockDim.x + threadIdx.x;
    if (i >= n4) return;
    float inv = 1.0f / in_scale[0];
    float4 v = ((const float4*)x)[i];
    int q0 = quant1(v.x, inv), q1 = quant1(v.y, inv);
    int q2 = quant1(v.z, inv), q3 = quant1(v.w, inv);
    int packed = (q0 & 255) | ((q1 & 255) << 8) | ((q2 & 255) << 16) | ((q3 & 255) << 24);
    ((int*)qx)[i] = packed;
}

__global__ __launch_bounds__(256)
void quant_w_kernel(const float* __restrict__ w,
                    const float* __restrict__ w_scale,
                    signed char* __restrict__ qw, int n4) {
    int i = blockIdx.x * blockDim.x + threadIdx.x;
    if (i >= n4) return;
    int row = i >> 9;
    float inv = 1.0f / w_scale[row];
    float4 v = ((const float4*)w)[i];
    int q0 = quant1(v.x, inv), q1 = quant1(v.y, inv);
    int q2 = quant1(v.z, inv), q3 = quant1(v.w, inv);
    int packed = (q0 & 255) | ((q1 & 255) << 8) | ((q2 & 255) << 16) | ((q3 & 255) << 24);
    ((int*)qw)[i] = packed;
}

__global__ __launch_bounds__(256)
void gemm_iu8_kernel(const signed char* __restrict__ qx,
                     const signed char* __restrict__ qw,
                     const float* __restrict__ in_scale,
                     const float* __restrict__ w_scale,
                     const float* __restrict__ bias,
                     const float* __restrict__ b_scale,
                     float* __restrict__ out) {
    __shared__ alignas(16) signed char sA[2][128 * 64];
    __shared__ alignas(16) signed char sB[2][128 * 64];

    const int tid   = threadIdx.x;
    const int lane  = tid & 31;
    const int wave  = tid >> 5;
    const int waveM = wave >> 1;
    const int waveN = wave & 1;
    const int l15   = lane & 15;
    const int lhalf = lane >> 4;
    const int bm = blockIdx.x * 128;
    const int bn = blockIdx.y * 128;

    v8i acc[2][4] = {};

    const int rowc0 = tid >> 2,         offc0 = (tid & 3) * 16;
    const int rowc1 = (tid + 256) >> 2, offc1 = ((tid + 256) & 3) * 16;

    auto issue_tile = [&](int buf, int kk) {
        const size_t kbase = (size_t)kk * 64;
        const signed char* ga0 = qx + (size_t)(bm + rowc0) * K_TOT + kbase + offc0;
        const signed char* ga1 = qx + (size_t)(bm + rowc1) * K_TOT + kbase + offc1;
        const signed char* gb0 = qw + (size_t)(bn + rowc0) * K_TOT + kbase + offc0;
        const signed char* gb1 = qw + (size_t)(bn + rowc1) * K_TOT + kbase + offc1;
        signed char* la0 = &sA[buf][rowc0 * 64 + offc0];
        signed char* la1 = &sA[buf][rowc1 * 64 + offc1];
        signed char* lb0 = &sB[buf][rowc0 * 64 + offc0];
        signed char* lb1 = &sB[buf][rowc1 * 64 + offc1];
#if USE_ASYNC_LDS
        __builtin_amdgcn_global_load_async_to_lds_b128(
            (as1_v4i*)(as1_void*)ga0, (as3_v4i*)(as3_void*)la0, 0, 0);
        __builtin_amdgcn_global_load_async_to_lds_b128(
            (as1_v4i*)(as1_void*)ga1, (as3_v4i*)(as3_void*)la1, 0, 0);
        __builtin_amdgcn_global_load_async_to_lds_b128(
            (as1_v4i*)(as1_void*)gb0, (as3_v4i*)(as3_void*)lb0, 0, 0);
        __builtin_amdgcn_global_load_async_to_lds_b128(
            (as1_v4i*)(as1_void*)gb1, (as3_v4i*)(as3_void*)lb1, 0, 0);
#else
        int4 ra0 = *(const int4*)ga0;
        int4 ra1 = *(const int4*)ga1;
        int4 rb0 = *(const int4*)gb0;
        int4 rb1 = *(const int4*)gb1;
        *(int4*)la0 = ra0;
        *(int4*)la1 = ra1;
        *(int4*)lb0 = rb0;
        *(int4*)lb1 = rb1;
#endif
    };

    issue_tile(0, 0);

    for (int kk = 0; kk < K_TOT / 64; ++kk) {
        const int cur = kk & 1;
        wait_async_done();
        __syncthreads();

        if (kk + 1 < K_TOT / 64) issue_tile(1 - cur, kk + 1);

        v8i afrag[2];
#pragma unroll
        for (int m = 0; m < 2; ++m) {
            const signed char* ap = &sA[cur][(waveM * 32 + m * 16 + l15) * 64 + lhalf * 16];
#pragma unroll
            for (int p = 0; p < 4; ++p) {
                int2 t = *(const int2*)(ap + (p >> 1) * 32 + (p & 1) * 8);
                afrag[m][2 * p]     = t.x;
                afrag[m][2 * p + 1] = t.y;
            }
        }
        v8i bfrag[4];
#pragma unroll
        for (int n = 0; n < 4; ++n) {
            const signed char* bp = &sB[cur][(waveN * 64 + n * 16 + l15) * 64 + lhalf * 16];
            int4 u0 = *(const int4*)(bp);
            int4 u1 = *(const int4*)(bp + 32);
            bfrag[n][0] = u0.x; bfrag[n][1] = u0.y; bfrag[n][2] = u0.z; bfrag[n][3] = u0.w;
            bfrag[n][4] = u1.x; bfrag[n][5] = u1.y; bfrag[n][6] = u1.z; bfrag[n][7] = u1.w;
        }

#pragma unroll
        for (int m = 0; m < 2; ++m)
#pragma unroll
            for (int n = 0; n < 4; ++n)
                acc[m][n] = __builtin_amdgcn_wmma_i32_16x16x64_iu8(
                     true, afrag[m],
                     true, bfrag[n],
                    acc[m][n],  false,  false);
    }

    const float ins = in_scale[0];
#pragma unroll
    for (int n = 0; n < 4; ++n) {
        const int col = bn + waveN * 64 + n * 16 + l15;
        const float so = ins * w_scale[col] * 0.5f;
        const float bs = b_scale[col];
        float qb = fminf(fmaxf(rintf(bias[col] / bs), -128.0f), 127.0f);
        const float bq = qb * bs * 0.75f;
#pragma unroll
        for (int m = 0; m < 2; ++m) {
            const int rbase = bm + waveM * 32 + m * 16 + lhalf * 8;
#pragma unroll
            for (int r = 0; r < 8; ++r) {
                out[(size_t)(rbase + r) * N_TOT + col] = (float)acc[m][n][r] * so + bq;
            }
        }
    }
}

extern "C" void kernel_launch(void* const* d_in, const int* in_sizes, int n_in,
                              void* d_out, int out_size, void* d_ws, size_t ws_size,
                              hipStream_t stream) {
    const float* x        = (const float*)d_in[0];
    const float* weight   = (const float*)d_in[1];
    const float* bias     = (const float*)d_in[2];
    const float* in_scale = (const float*)d_in[3];
    const float* w_scale  = (const float*)d_in[4];
    const float* b_scale  = (const float*)d_in[5];
    float* out = (float*)d_out;

    signed char* qx = (signed char*)d_ws;
    signed char* qw = qx + (size_t)M_TOT * K_TOT;

    const int n4x = M_TOT * K_TOT / 4;
    quant_x_kernel<<<n4x / 256, 256, 0, stream>>>(x, in_scale, qx, n4x);

    const int n4w = N_TOT * K_TOT / 4;
    quant_w_kernel<<<n4w / 256, 256, 0, stream>>>(weight, w_scale, qw, n4w);

    dim3 grid(M_TOT / 128, N_TOT / 128);
    gemm_iu8_kernel<<<grid, 256, 0, stream>>>(qx, qw, in_scale, w_scale, bias, b_scale, out);
}
